// LocalContrastiveLoss_33457795235870
// MI455X (gfx1250) — matrix operand order repaired and checked by reading, not compiled, not run on hardware
//
#include <hip/hip_runtime.h>
#include <hip/hip_bf16.h>
#include <math.h>

typedef __attribute__((ext_vector_type(16))) _Float16 v16h;
typedef __attribute__((ext_vector_type(8)))  _Float16 v8h;
typedef __attribute__((ext_vector_type(8)))  float    v8f;

#define T_DIM 4096
#define B_DIM 16
#define D_DIM 128
#define TILE  16
#define BAND  48                 // rows t0-16 .. t0+31
#define INV_TEMP 10.0f
#define OUT_SCALE (1.0f / 131072.0f)   // 1 / (2 * T * B)

__global__ void lcl_init_out(float* out) {
    if (threadIdx.x < 2) out[threadIdx.x] = 0.0f;
}

__launch_bounds__(64)
__global__ void LocalContrastiveLoss_kernel(const float* __restrict__ mfeat,
                                            const float* __restrict__ afeat,
                                            float* __restrict__ out)
{
    // LDS: normalized f16 band rows for both tensors + f32 logit staging
    __shared__ __attribute__((aligned(32))) _Float16 mb[BAND * D_DIM]; // 12 KB
    __shared__ __attribute__((aligned(32))) _Float16 ab[BAND * D_DIM]; // 12 KB
    __shared__ float lg[2 * 3 * 256];                                  // 6 KB

    const int t0   = blockIdx.x * TILE;
    const int b    = blockIdx.y;
    const int tid  = threadIdx.x;
    const int wave = tid >> 5;     // 0 or 1 (wave32)
    const int lane = tid & 31;

    // ---------------- Phase 1: load + L2-normalize band rows (f32 -> f16 LDS)
    {
        const float* src = (wave == 0) ? mfeat : afeat;
        _Float16*    dst = (wave == 0) ? mb    : ab;
        for (int br = 0; br < BAND; ++br) {
            int g = t0 - 16 + br;
            g = g < 0 ? 0 : (g > T_DIM - 1 ? T_DIM - 1 : g);   // clamp (masked later)
            const float* row = src + ((size_t)b * T_DIM + g) * D_DIM;
            float4 v = ((const float4*)row)[lane];             // 32 lanes x 4 = 128
            float ss = v.x * v.x + v.y * v.y + v.z * v.z + v.w * v.w;
            #pragma unroll
            for (int s = 16; s > 0; s >>= 1) ss += __shfl_xor(ss, s, 32);
            float inv = 1.0f / fmaxf(sqrtf(ss), 1e-12f);
            _Float16* drow = dst + br * D_DIM + lane * 4;
            drow[0] = (_Float16)(v.x * inv);
            drow[1] = (_Float16)(v.y * inv);
            drow[2] = (_Float16)(v.z * inv);
            drow[3] = (_Float16)(v.w * inv);
        }
    }
    __syncthreads();

    // ---------------- Phase 2: banded GEMM via WMMA. wave == direction.
    // dir0: q = m (center tile), k = a band. dir1: q = a, k = m.
    {
        const _Float16* qb = (wave == 0) ? mb : ab;
        const _Float16* kb = (wave == 0) ? ab : mb;
        const int n  = lane & 15;
        const int lo = (lane < 16) ? 1 : 0;

        for (int c = 0; c < 3; ++c) {        // 3 key tiles: j = t0-16+16c+n
            v8f acc = {};
            #pragma unroll
            for (int s = 0; s < 4; ++s) {    // D reduction: 4 x K=32
                // A (16x32 f16): lane<16 holds K s*32+0..15 of row M=n,
                //                lane>=16 holds K s*32+16..31
                const _Float16* qrow = qb + (16 + n) * D_DIM + s * 32 + (lo ? 0 : 16);
                union { v16h v; v8h h[2]; } A;
                A.h[0] = *(const v8h*)(qrow);
                A.h[1] = *(const v8h*)(qrow + 8);
                // B (32x16 f16): column j == key row (16c+n);
                //                lane<16: K s*32+0..15, lane>=16: K s*32+16..31
                const _Float16* krow = kb + (16 * c + n) * D_DIM + s * 32 + (lo ? 0 : 16);
                v16h Bm = *(const v16h*)(krow);
                acc = __builtin_amdgcn_wmma_f32_16x16x32_f16(
                        false, A.v, false, Bm, (short)0, acc, false, false);
            }
            // C/D layout: VGPR r, lanes 0-15 -> (M=r,   N=lane)
            //                     lanes16-31 -> (M=r+8, N=lane-16)
            float* dstl = lg + (wave * 3 + c) * 256 + (lo ? 0 : 8 * 16) + n;
            #pragma unroll
            for (int r = 0; r < 8; ++r) dstl[r * 16] = acc[r];
        }
    }
    __syncthreads();

    // ---------------- Phase 3: masked LSE + argmax per (dir, query row)
    if (wave == 0) {
        const int dir = lane >> 4;
        const int M   = lane & 15;
        const int t   = t0 + M;
        const float* base = lg + dir * 3 * 256;

        float max_all = -1e30f, max_pos = -1e30f;
        for (int c = 0; c < 3; ++c)
            for (int nn = 0; nn < 16; ++nn) {
                int j = t0 - 16 + 16 * c + nn;
                int off = j - t;
                if (off < -16 || off >= 16 || j < 0 || j >= T_DIM) continue;
                float l = base[c * 256 + M * 16 + nn] * INV_TEMP;
                max_all = fmaxf(max_all, l);
                if (off >= -4 && off < 4) max_pos = fmaxf(max_pos, l);
            }
        float sum_all = 0.0f, sum_pos = 0.0f;
        for (int c = 0; c < 3; ++c)
            for (int nn = 0; nn < 16; ++nn) {
                int j = t0 - 16 + 16 * c + nn;
                int off = j - t;
                if (off < -16 || off >= 16 || j < 0 || j >= T_DIM) continue;
                float l = base[c * 256 + M * 16 + nn] * INV_TEMP;
                sum_all += __expf(l - max_all);
                if (off >= -4 && off < 4) sum_pos += __expf(l - max_pos);
            }
        float loss    = (max_all + __logf(sum_all)) - (max_pos + __logf(sum_pos));
        float correct = (max_pos >= max_all) ? 1.0f : 0.0f;

        #pragma unroll
        for (int s = 16; s > 0; s >>= 1) {
            loss    += __shfl_xor(loss,    s, 32);
            correct += __shfl_xor(correct, s, 32);
        }
        if (lane == 0) {
            atomicAdd(&out[0], loss    * OUT_SCALE);
            atomicAdd(&out[1], correct * OUT_SCALE);
        }
    }
}

extern "C" void kernel_launch(void* const* d_in, const int* in_sizes, int n_in,
                              void* d_out, int out_size, void* d_ws, size_t ws_size,
                              hipStream_t stream) {
    (void)in_sizes; (void)n_in; (void)out_size; (void)d_ws; (void)ws_size;
    const float* mfeat = (const float*)d_in[0];
    const float* afeat = (const float*)d_in[1];
    float* out = (float*)d_out;

    lcl_init_out<<<1, 32, 0, stream>>>(out);
    dim3 grid(T_DIM / TILE, B_DIM);   // (256, 16) workgroups
    LocalContrastiveLoss_kernel<<<grid, 64, 0, stream>>>(mfeat, afeat, out);
}
